// SymbolicDiffusion_54666343744033
// MI455X (gfx1250) — matrix operand order repaired and checked by reading, not compiled, not run on hardware
//
#include <hip/hip_runtime.h>
#include <math.h>

// ---------------------------------------------------------------------------
// Sizes
// ---------------------------------------------------------------------------
#define BB   64
#define LL   256
#define DD   512
#define HH   4
#define HD   128
#define VV   2048
#define TT   1000
#define NPTS 250
#define NCIN 2
#define NLAYER 4
#define EPS  1e-5f

typedef float    v8f  __attribute__((ext_vector_type(8)));
typedef _Float16 v16h __attribute__((ext_vector_type(16)));
typedef _Float16 h8   __attribute__((ext_vector_type(8)));

union Frag16 { v16h v; h8 h[2]; };

// ---------------------------------------------------------------------------
// WMMA GEMM:  C[z][M,N] = A[z][M,K] * B[z][N,K]^T  (+bias, activation)
//   A,B f16 row-major (lda=K, ldb=K), C f32 and/or f16 (ldc=N)
//   act: 0=none 1=relu 2=exact gelu
// Tile: 128x128x32, 256 threads (8 waves), each wave 64x32 via 4x2 WMMA tiles.
// Global->LDS staging uses CDNA5 GLOBAL_LOAD_ASYNC_TO_LDS_B128 (ASYNCcnt)
// with a 2-stage LDS double buffer so loads overlap WMMA compute.
// ---------------------------------------------------------------------------
#define GBM 128
#define GBN 128
#define GBK 32
#define LDSW 40  // padded row stride in halves (80B -> 16B aligned rows)

__global__ __launch_bounds__(256, 2)
void k_gemm_nt(const _Float16* __restrict__ A, const _Float16* __restrict__ B,
               const float* __restrict__ bias,
               float* __restrict__ Cf, _Float16* __restrict__ Ch,
               int M, int N, int K,
               long batchA, long batchB, long batchCf, long batchCh,
               int act)
{
    __shared__ _Float16 As[2][GBM * LDSW];
    __shared__ _Float16 Bs[2][GBN * LDSW];

    const int z = blockIdx.z;
    A += (long)z * batchA;
    B += (long)z * batchB;
    if (Cf) Cf += (long)z * batchCf;
    if (Ch) Ch += (long)z * batchCh;

    const int bm   = blockIdx.y * GBM;
    const int bn   = blockIdx.x * GBN;
    const int tid  = threadIdx.x;
    const int lane = tid & 31;
    const int wave = tid >> 5;
    const int wm   = (wave & 1) * 64;   // wave M offset in tile
    const int wn   = (wave >> 1) * 32;  // wave N offset in tile

    const int lr = tid >> 1;            // 0..127 : staged row
    const int lc = (tid & 1) * 16;      // 0 / 16 : staged col seg

    v8f acc[4][2] = {};

    const int mrow = lane & 15;
    const int ksel = lane >> 4;         // 0/1

    // per-thread async staging of one 32-byte row segment of A and of B
    auto stage = [&](int buf, int k0) {
        const int gr = bm + lr;
        const unsigned ldsA = (unsigned)(unsigned long)&As[buf][lr * LDSW + lc];
        const unsigned ldsB = (unsigned)(unsigned long)&Bs[buf][lr * LDSW + lc];
        if (gr < M) {
            const unsigned long ga = (unsigned long)(A + (long)gr * K + k0 + lc);
            asm volatile(
                "global_load_async_to_lds_b128 %0, %1, off\n\t"
                "global_load_async_to_lds_b128 %0, %1, off offset:16"
                :: "v"(ldsA), "v"(ga) : "memory");
        }
        const unsigned long gb = (unsigned long)(B + (long)(bn + lr) * K + k0 + lc);
        asm volatile(
            "global_load_async_to_lds_b128 %0, %1, off\n\t"
            "global_load_async_to_lds_b128 %0, %1, off offset:16"
            :: "v"(ldsB), "v"(gb) : "memory");
    };

    const int nk = K / GBK;

    stage(0, 0);
    asm volatile("s_wait_asynccnt 0" ::: "memory");
    __syncthreads();

    for (int kt = 0; kt < nk; ++kt) {
        const int cur = kt & 1;

        // issue async loads for the next K tile into the other buffer
        if (kt + 1 < nk) stage(1 - cur, (kt + 1) * GBK);

        // prefetch tile kt+2 toward L2 so the next async load hits cache
        if (kt + 2 < nk) {
            const int gr = bm + lr;
            const int kp = (kt + 2) * GBK;
            if (gr < M) __builtin_prefetch(A + (long)gr * K + kp + lc, 0, 0);
            __builtin_prefetch(B + (long)(bn + lr) * K + kp + lc, 0, 0);
        }

        // ---- load fragments per ISA 16-bit A/B VGPR layouts ----
        Frag16 af[4], bf[2];
#pragma unroll
        for (int i = 0; i < 4; ++i) {
            const _Float16* p = &As[cur][(wm + i * 16 + mrow) * LDSW];
            af[i].h[0] = *(const h8*)(p + ksel * 8);        // K 0..7 / 8..15
            af[i].h[1] = *(const h8*)(p + ksel * 8 + 16);   // K 16..23 / 24..31
        }
#pragma unroll
        for (int j = 0; j < 2; ++j) {
            const _Float16* p = &Bs[cur][(wn + j * 16 + mrow) * LDSW + ksel * 8];
            bf[j].h[0] = *(const h8*)(p);
            bf[j].h[1] = *(const h8*)(p + 16);
        }

#pragma unroll
        for (int i = 0; i < 4; ++i)
#pragma unroll
            for (int j = 0; j < 2; ++j)
                acc[i][j] = __builtin_amdgcn_wmma_f32_16x16x32_f16(
                    false, af[i].v, false, bf[j].v,
                    (short)0, acc[i][j], false, false);

        // next buffer must be fully resident before the next iteration
        asm volatile("s_wait_asynccnt 0" ::: "memory");
        __syncthreads();
    }

    // ---- epilogue: C/D layout (lane = n + 16*hi, vgpr v -> m = v + 8*hi) ----
    const int nlo = lane & 15;
    const int hi  = lane >> 4;
#pragma unroll
    for (int i = 0; i < 4; ++i) {
#pragma unroll
        for (int j = 0; j < 2; ++j) {
#pragma unroll
            for (int v = 0; v < 8; ++v) {
                const int m = bm + wm + i * 16 + v + 8 * hi;
                const int n = bn + wn + j * 16 + nlo;
                if (m < M) {
                    float val = acc[i][j][v];
                    if (bias) val += bias[n];
                    if (act == 1)      val = fmaxf(val, 0.f);
                    else if (act == 2) val = 0.5f * val * (1.f + erff(val * 0.70710678118654752f));
                    if (Cf) Cf[(long)m * N + n] = val;
                    if (Ch) Ch[(long)m * N + n] = (_Float16)val;
                }
            }
        }
    }
}

// ---------------------------------------------------------------------------
// Elementwise / reduction helpers
// ---------------------------------------------------------------------------
__global__ void k_f16cvt(const float* __restrict__ in, _Float16* __restrict__ out, long n)
{
    long i = (long)blockIdx.x * blockDim.x + threadIdx.x;
    if (i < n) out[i] = (_Float16)in[i];
}

// per-channel stats over rows, x layout [rows, C] (channels-last)
__global__ void k_bn_stats(const float* __restrict__ x, int rows, int C,
                           float* __restrict__ mean, float* __restrict__ rstd)
{
    const int c = blockIdx.x;
    float s = 0.f, s2 = 0.f;
    for (int r = threadIdx.x; r < rows; r += blockDim.x) {
        float v = x[(long)r * C + c];
        s += v; s2 += v * v;
    }
    __shared__ float sh[512];
    sh[threadIdx.x] = s; sh[threadIdx.x + 256] = s2;
    __syncthreads();
    for (int o = 128; o > 0; o >>= 1) {
        if (threadIdx.x < o) {
            sh[threadIdx.x]       += sh[threadIdx.x + o];
            sh[threadIdx.x + 256] += sh[threadIdx.x + 256 + o];
        }
        __syncthreads();
    }
    if (threadIdx.x == 0) {
        float m   = sh[0] / (float)rows;
        float var = sh[256] / (float)rows - m * m;
        mean[c] = m;
        rstd[c] = rsqrtf(var + EPS);
    }
}

__global__ void k_bn_apply(const float* __restrict__ x,
                           const float* __restrict__ mean, const float* __restrict__ rstd,
                           const float* __restrict__ g, const float* __restrict__ b,
                           long total, int C,
                           _Float16* __restrict__ outH, float* __restrict__ outF, int relu)
{
    long i = (long)blockIdx.x * blockDim.x + threadIdx.x;
    if (i >= total) return;
    int c = (int)(i % C);
    float v = (x[i] - mean[c]) * rstd[c] * g[c] + b[c];
    if (relu) v = fmaxf(v, 0.f);
    if (outH) outH[i] = (_Float16)v;
    if (outF) outF[i] = v;
}

// stats of input points [B, 2, NPTS] per channel
__global__ void k_points_stats(const float* __restrict__ pts,
                               float* __restrict__ mean, float* __restrict__ rstd)
{
    const int c = blockIdx.x;
    float s = 0.f, s2 = 0.f;
    for (int i = threadIdx.x; i < BB * NPTS; i += blockDim.x) {
        int b = i / NPTS, n = i % NPTS;
        float v = pts[(b * NCIN + c) * NPTS + n];
        s += v; s2 += v * v;
    }
    __shared__ float sh[512];
    sh[threadIdx.x] = s; sh[threadIdx.x + 256] = s2;
    __syncthreads();
    for (int o = 128; o > 0; o >>= 1) {
        if (threadIdx.x < o) {
            sh[threadIdx.x]       += sh[threadIdx.x + o];
            sh[threadIdx.x + 256] += sh[threadIdx.x + 256 + o];
        }
        __syncthreads();
    }
    if (threadIdx.x == 0) {
        const float rows = (float)(BB * NPTS);
        float m   = sh[0] / rows;
        float var = sh[256] / rows - m * m;
        mean[c] = m;
        rstd[c] = rsqrtf(var + EPS);
    }
}

// conv1 (K=2): out[r, o] = sum_c bn_in(pts)[r,c] * w[o,c] + b[o]   (channels-last rows)
__global__ void k_c1(const float* __restrict__ pts,
                     const float* __restrict__ mean, const float* __restrict__ rstd,
                     const float* __restrict__ gin, const float* __restrict__ bin,
                     const float* __restrict__ w, const float* __restrict__ bb,
                     float* __restrict__ out)
{
    long i = (long)blockIdx.x * blockDim.x + threadIdx.x;
    const long total = (long)BB * NPTS * DD;
    if (i >= total) return;
    int o = (int)(i % DD);
    long r = i / DD;
    int b = (int)(r / NPTS), n = (int)(r % NPTS);
    float acc = bb[o];
#pragma unroll
    for (int c = 0; c < NCIN; ++c) {
        float v = pts[(b * NCIN + c) * NPTS + n];
        v = (v - mean[c]) * rstd[c] * gin[c] + bin[c];
        acc += v * w[o * NCIN + c];
    }
    out[i] = acc;
}

// max over points: h [B*NPTS, 2048] f16 -> out [B, 2048] f16
__global__ void k_maxpool(const _Float16* __restrict__ h, _Float16* __restrict__ out)
{
    long i = (long)blockIdx.x * blockDim.x + threadIdx.x;
    if (i >= (long)BB * 2048) return;
    int c = (int)(i % 2048), b = (int)(i / 2048);
    float m = -3.0e38f;
    for (int n = 0; n < NPTS; ++n)
        m = fmaxf(m, (float)h[((long)(b * NPTS + n)) * 2048 + c]);
    out[i] = (_Float16)m;
}

__global__ void k_cond(const float* __restrict__ tnet, const float* __restrict__ vemb,
                       const int* __restrict__ vars, const int* __restrict__ uncond,
                       float* __restrict__ cond)
{
    long i = (long)blockIdx.x * blockDim.x + threadIdx.x;
    if (i >= (long)BB * DD) return;
    int d = (int)(i % DD), b = (int)(i / DD);
    float ve = vemb[(long)vars[b] * DD + d];
    cond[i] = uncond[b] ? ve : (tnet[i] + ve);
}

// diffusion schedule per batch element
__global__ void k_schedule(const int* __restrict__ t, float* __restrict__ sched)
{
    int b = threadIdx.x;
    if (b >= BB) return;
    int tb = t[b];
    float abar = 1.f, beta_t = 0.f;
    for (int i = 0; i <= tb; ++i) {
        float beta = 1e-4f + (0.02f - 1e-4f) * ((float)i / (float)(TT - 1));
        abar *= (1.f - beta);
        if (i == tb) beta_t = beta;
    }
    sched[b * 4 + 0] = sqrtf(abar);
    sched[b * 4 + 1] = sqrtf(1.f - abar);
    sched[b * 4 + 2] = beta_t;
    sched[b * 4 + 3] = 1.f - beta_t;   // alpha_t
}

__global__ void k_build_x(const int* __restrict__ tokens, const float* __restrict__ tok_emb,
                          const float* __restrict__ pos, const float* __restrict__ time_emb,
                          const int* __restrict__ t, const float* __restrict__ cond,
                          const float* __restrict__ noise, const float* __restrict__ sched,
                          float* __restrict__ xt, float* __restrict__ x,
                          _Float16* __restrict__ xh)
{
    long i = (long)blockIdx.x * blockDim.x + threadIdx.x;
    const long total = (long)BB * LL * DD;
    if (i >= total) return;
    int d = (int)(i % DD);
    long bl = i / DD;
    int l = (int)(bl % LL), b = (int)(bl / LL);
    int tok = tokens[b * LL + l];
    float x0 = (tok == 0) ? 0.f : tok_emb[(long)tok * DD + d];  // padding_idx=0
    float xtv = sched[b * 4] * x0 + sched[b * 4 + 1] * noise[i];
    float xv  = xtv + pos[l * DD + d] + time_emb[(long)t[b] * DD + d] + cond[b * DD + d];
    xt[i] = xtv; x[i] = xv; xh[i] = (_Float16)xv;
}

// qkv [B,L,3D] f16 -> Q,K [BH, L, HD] ; Vt [BH, HD, L]
__global__ void k_split_heads(const _Float16* __restrict__ qkv,
                              _Float16* __restrict__ Q, _Float16* __restrict__ K2,
                              _Float16* __restrict__ Vt)
{
    long i = (long)blockIdx.x * blockDim.x + threadIdx.x;
    const long total = (long)BB * HH * LL * HD;
    if (i >= total) return;
    int d = (int)(i % HD);
    long r = i / HD;
    int l = (int)(r % LL);
    long z = r / LL;                 // b*H + h
    int h = (int)(z % HH), b = (int)(z / HH);
    const _Float16* src = qkv + ((long)(b * LL + l)) * (3 * DD) + h * HD + d;
    Q[i]  = src[0];
    K2[i] = src[DD];
    Vt[(z * HD + d) * LL + l] = src[2 * DD];
}

// softmax row with 1/sqrt(HD) scale: grid(L, BH), block 256
__global__ void k_softmax(const float* __restrict__ S, _Float16* __restrict__ P)
{
    const long base = ((long)blockIdx.y * LL + blockIdx.x) * LL;
    const int tx = threadIdx.x;
    float v = S[base + tx] * 0.08838834764831845f;
    __shared__ float sh[256];
    sh[tx] = v; __syncthreads();
    for (int o = 128; o > 0; o >>= 1) {
        if (tx < o) sh[tx] = fmaxf(sh[tx], sh[tx + o]);
        __syncthreads();
    }
    float mx = sh[0]; __syncthreads();
    float e = __expf(v - mx);
    sh[tx] = e; __syncthreads();
    for (int o = 128; o > 0; o >>= 1) {
        if (tx < o) sh[tx] += sh[tx + o];
        __syncthreads();
    }
    P[base + tx] = (_Float16)(e / sh[0]);
}

// Oh [BH, L, HD] -> Om [B, L, D]
__global__ void k_merge_heads(const _Float16* __restrict__ Oh, _Float16* __restrict__ Om)
{
    long i = (long)blockIdx.x * blockDim.x + threadIdx.x;
    const long total = (long)BB * LL * DD;
    if (i >= total) return;
    int d = (int)(i % DD);
    long bl = i / DD;
    int l = (int)(bl % LL), b = (int)(bl / LL);
    int h = d / HD, dd = d % HD;
    Om[i] = Oh[(((long)(b * HH + h) * LL + l) * HD + dd)];
}

// x = LN(x + o);   one block (256 thr) per row of 512
__global__ void k_add_ln(float* __restrict__ x, const float* __restrict__ o,
                         const float* __restrict__ g, const float* __restrict__ b,
                         _Float16* __restrict__ xh)
{
    const long r = blockIdx.x;
    const int tx = threadIdx.x;
    float v0 = x[r * DD + tx]       + o[r * DD + tx];
    float v1 = x[r * DD + 256 + tx] + o[r * DD + 256 + tx];
    __shared__ float sh[256];
    sh[tx] = v0 + v1; __syncthreads();
    for (int s = 128; s > 0; s >>= 1) {
        if (tx < s) sh[tx] += sh[tx + s];
        __syncthreads();
    }
    float m = sh[0] / (float)DD; __syncthreads();
    float d0 = v0 - m, d1 = v1 - m;
    sh[tx] = d0 * d0 + d1 * d1; __syncthreads();
    for (int s = 128; s > 0; s >>= 1) {
        if (tx < s) sh[tx] += sh[tx + s];
        __syncthreads();
    }
    float rs = rsqrtf(sh[0] / (float)DD + EPS);
    float y0 = g[tx] * d0 * rs + b[tx];
    float y1 = g[tx + 256] * d1 * rs + b[tx + 256];
    x[r * DD + tx] = y0;        x[r * DD + 256 + tx] = y1;
    xh[r * DD + tx] = (_Float16)y0;
    xh[r * DD + 256 + tx] = (_Float16)y1;
}

// p_sample mean + copy outputs
__global__ void k_mean(const float* __restrict__ xt, const float* __restrict__ np_,
                       const float* __restrict__ noise, const float* __restrict__ sched,
                       float* __restrict__ out_np, float* __restrict__ out_noise,
                       _Float16* __restrict__ meanH)
{
    long i = (long)blockIdx.x * blockDim.x + threadIdx.x;
    const long total = (long)BB * LL * DD;
    if (i >= total) return;
    int b = (int)(i / ((long)LL * DD));
    float v = np_[i];
    out_np[i] = v;
    out_noise[i] = noise[i];
    float s1 = sched[b * 4 + 1], beta = sched[b * 4 + 2], alpha = sched[b * 4 + 3];
    meanH[i] = (_Float16)((xt[i] - beta / s1 * v) * rsqrtf(alpha));
}

// ---------------------------------------------------------------------------
// Host
// ---------------------------------------------------------------------------
enum {
    IN_POINTS = 0, IN_TOKENS, IN_VARS, IN_T, IN_NOISE, IN_UNCOND,
    P_TOK_EMB, P_VARS_EMB, P_POS_EMB, P_TIME_EMB,
    TN_BNIN_G, TN_BNIN_B, TN_C1_W, TN_C1_B, TN_BN1_G, TN_BN1_B,
    TN_C2_W, TN_C2_B, TN_BN2_G, TN_BN2_B,
    TN_C3_W, TN_C3_B, TN_BN3_G, TN_BN3_B,
    TN_FC1_W, TN_FC1_B, TN_BN4_G, TN_BN4_B,
    TN_FC2_W, TN_FC2_B, TN_BN5_G, TN_BN5_B,
    L_BASE = 32,                 // 12 entries per layer
    P_DEC_W = L_BASE + 12 * NLAYER, P_DEC_B
};
// per-layer: 0 in_w, 1 in_b, 2 out_w, 3 out_b, 4 ln1_g, 5 ln1_b,
//            6 ff1_w, 7 ff1_b, 8 ff2_w, 9 ff2_b, 10 ln2_g, 11 ln2_b

static inline void gemm_nt(hipStream_t s, const _Float16* A, const _Float16* B,
                           const float* bias, float* Cf, _Float16* Ch,
                           int M, int N, int K, int Z,
                           long bA, long bB, long bCf, long bCh, int act)
{
    dim3 grid(N / GBN, (M + GBM - 1) / GBM, Z);
    k_gemm_nt<<<grid, 256, 0, s>>>(A, B, bias, Cf, Ch, M, N, K, bA, bB, bCf, bCh, act);
}

extern "C" void kernel_launch(void* const* d_in, const int* in_sizes, int n_in,
                              void* d_out, int out_size, void* d_ws, size_t ws_size,
                              hipStream_t stream)
{
    (void)in_sizes; (void)n_in; (void)out_size; (void)ws_size;

    const float* PTS     = (const float*)d_in[IN_POINTS];
    const int*   TOKENS  = (const int*)  d_in[IN_TOKENS];
    const int*   VARS    = (const int*)  d_in[IN_VARS];
    const int*   TVEC    = (const int*)  d_in[IN_T];
    const float* NOISE   = (const float*)d_in[IN_NOISE];
    const int*   UNCOND  = (const int*)  d_in[IN_UNCOND];
    const float* TOK_EMB = (const float*)d_in[P_TOK_EMB];
    const float* VEMB    = (const float*)d_in[P_VARS_EMB];
    const float* POS     = (const float*)d_in[P_POS_EMB];
    const float* TIME_E  = (const float*)d_in[P_TIME_EMB];
    auto F = [&](int i) { return (const float*)d_in[i]; };

    // ---- workspace carve-up ----
    char* wsb = (char*)d_ws;
    size_t off = 0;
    auto alloc = [&](size_t bytes) -> char* {
        char* p = wsb + off;
        off = (off + bytes + 255) & ~(size_t)255;
        return p;
    };
    const long ROWS_PT = (long)BB * NPTS;              // 16000
    const long ROWS_X  = (long)BB * LL;                // 16384
    const long BHL     = (long)BB * HH * LL;           // 65536

    _Float16* whC2  = (_Float16*)alloc(1024 * 512 * 2);
    _Float16* whC3  = (_Float16*)alloc(2048 * 1024 * 2);
    _Float16* whFc1 = (_Float16*)alloc(1024 * 2048 * 2);
    _Float16* whFc2 = (_Float16*)alloc(512 * 1024 * 2);
    _Float16* whIn  = (_Float16*)alloc((size_t)NLAYER * 1536 * 512 * 2);
    _Float16* whOut = (_Float16*)alloc((size_t)NLAYER * 512 * 512 * 2);
    _Float16* whF1  = (_Float16*)alloc((size_t)NLAYER * 2048 * 512 * 2);
    _Float16* whF2  = (_Float16*)alloc((size_t)NLAYER * 512 * 2048 * 2);
    _Float16* whDec = (_Float16*)alloc(2048 * 512 * 2);

    float* statM  = (float*)alloc(2048 * 4);
    float* statR  = (float*)alloc(2048 * 4);
    float* sched  = (float*)alloc(BB * 4 * 4);
    float* condF  = (float*)alloc((size_t)BB * DD * 4);
    float* tnetF  = (float*)alloc((size_t)BB * DD * 4);
    _Float16* pooled = (_Float16*)alloc((size_t)BB * 2048 * 2);
    float* gF     = (float*)alloc((size_t)BB * 2048 * 4);       // fc intermediates
    _Float16* gH  = (_Float16*)alloc((size_t)BB * 2048 * 2);

    float*    actF = (float*)   alloc(ROWS_PT * 2048 * 4);      // 131MB: h1/h2/h3, scores, attn, ff2
    _Float16* actH = (_Float16*)alloc(ROWS_PT * 2048 * 2);      // 65.5MB: h f16, ff hidden

    float*    xtF  = (float*)   alloc(ROWS_X * DD * 4);
    float*    xF   = (float*)   alloc(ROWS_X * DD * 4);
    _Float16* xh   = (_Float16*)alloc(ROWS_X * DD * 2);
    _Float16* qkvh = (_Float16*)alloc(ROWS_X * 3 * DD * 2);
    _Float16* Qh   = (_Float16*)alloc(BHL * HD * 2);
    _Float16* Kh   = (_Float16*)alloc(BHL * HD * 2);
    _Float16* Vt   = (_Float16*)alloc(BHL * HD * 2);
    _Float16* Oh   = (_Float16*)alloc(BHL * HD * 2);
    _Float16* Om   = (_Float16*)alloc(ROWS_X * DD * 2);
    _Float16* prbH = (_Float16*)alloc(BHL * LL * 2);
    _Float16* meanH= (_Float16*)alloc(ROWS_X * DD * 2);

    float* yPred   = (float*)d_out;
    float* outNP   = yPred + ROWS_X * (long)VV;
    float* outNoise= outNP + ROWS_X * (long)DD;

    auto cvt = [&](const float* src, _Float16* dst, long n) {
        k_f16cvt<<<dim3((unsigned)((n + 255) / 256)), 256, 0, stream>>>(src, dst, n);
    };
    auto ew = [&](long n) { return dim3((unsigned)((n + 255) / 256)); };

    // ---- convert weights to f16 ----
    cvt(F(TN_C2_W),  whC2,  1024L * 512);
    cvt(F(TN_C3_W),  whC3,  2048L * 1024);
    cvt(F(TN_FC1_W), whFc1, 1024L * 2048);
    cvt(F(TN_FC2_W), whFc2, 512L * 1024);
    for (int l = 0; l < NLAYER; ++l) {
        cvt(F(L_BASE + 12 * l + 0), whIn  + (long)l * 1536 * 512, 1536L * 512);
        cvt(F(L_BASE + 12 * l + 2), whOut + (long)l * 512 * 512,  512L * 512);
        cvt(F(L_BASE + 12 * l + 6), whF1  + (long)l * 2048 * 512, 2048L * 512);
        cvt(F(L_BASE + 12 * l + 8), whF2  + (long)l * 512 * 2048, 512L * 2048);
    }
    cvt(F(P_DEC_W), whDec, 2048L * 512);

    // ================= tnet =================
    k_points_stats<<<dim3(NCIN), 256, 0, stream>>>(PTS, statM, statR);
    k_c1<<<ew(ROWS_PT * DD), 256, 0, stream>>>(PTS, statM, statR,
        F(TN_BNIN_G), F(TN_BNIN_B), F(TN_C1_W), F(TN_C1_B), actF);
    k_bn_stats<<<dim3(DD), 256, 0, stream>>>(actF, (int)ROWS_PT, DD, statM, statR);
    k_bn_apply<<<ew(ROWS_PT * DD), 256, 0, stream>>>(actF, statM, statR,
        F(TN_BN1_G), F(TN_BN1_B), ROWS_PT * DD, DD, actH, nullptr, 1);

    gemm_nt(stream, actH, whC2, F(TN_C2_B), actF, nullptr,
            (int)ROWS_PT, 1024, 512, 1, 0, 0, 0, 0, 0);
    k_bn_stats<<<dim3(1024), 256, 0, stream>>>(actF, (int)ROWS_PT, 1024, statM, statR);
    k_bn_apply<<<ew(ROWS_PT * 1024), 256, 0, stream>>>(actF, statM, statR,
        F(TN_BN2_G), F(TN_BN2_B), ROWS_PT * 1024, 1024, actH, nullptr, 1);

    gemm_nt(stream, actH, whC3, F(TN_C3_B), actF, nullptr,
            (int)ROWS_PT, 2048, 1024, 1, 0, 0, 0, 0, 0);
    k_bn_stats<<<dim3(2048), 256, 0, stream>>>(actF, (int)ROWS_PT, 2048, statM, statR);
    k_bn_apply<<<ew(ROWS_PT * 2048), 256, 0, stream>>>(actF, statM, statR,
        F(TN_BN3_G), F(TN_BN3_B), ROWS_PT * 2048, 2048, actH, nullptr, 1);

    k_maxpool<<<ew((long)BB * 2048), 256, 0, stream>>>(actH, pooled);

    gemm_nt(stream, pooled, whFc1, F(TN_FC1_B), gF, nullptr,
            BB, 1024, 2048, 1, 0, 0, 0, 0, 0);
    k_bn_stats<<<dim3(1024), 256, 0, stream>>>(gF, BB, 1024, statM, statR);
    k_bn_apply<<<ew((long)BB * 1024), 256, 0, stream>>>(gF, statM, statR,
        F(TN_BN4_G), F(TN_BN4_B), (long)BB * 1024, 1024, gH, nullptr, 1);

    gemm_nt(stream, gH, whFc2, F(TN_FC2_B), gF, nullptr,
            BB, 512, 1024, 1, 0, 0, 0, 0, 0);
    k_bn_stats<<<dim3(512), 256, 0, stream>>>(gF, BB, 512, statM, statR);
    k_bn_apply<<<ew((long)BB * 512), 256, 0, stream>>>(gF, statM, statR,
        F(TN_BN5_G), F(TN_BN5_B), (long)BB * 512, 512, nullptr, tnetF, 1);

    k_cond<<<ew((long)BB * DD), 256, 0, stream>>>(tnetF, VEMB, VARS, UNCOND, condF);

    // ================= diffusion input =================
    k_schedule<<<dim3(1), 64, 0, stream>>>(TVEC, sched);
    k_build_x<<<ew(ROWS_X * DD), 256, 0, stream>>>(TOKENS, TOK_EMB, POS, TIME_E,
        TVEC, condF, NOISE, sched, xtF, xF, xh);

    // ================= transformer layers =================
    for (int l = 0; l < NLAYER; ++l) {
        const _Float16* wIn  = whIn  + (long)l * 1536 * 512;
        const _Float16* wOut = whOut + (long)l * 512 * 512;
        const _Float16* wF1  = whF1  + (long)l * 2048 * 512;
        const _Float16* wF2  = whF2  + (long)l * 512 * 2048;

        // qkv
        gemm_nt(stream, xh, wIn, F(L_BASE + 12 * l + 1), nullptr, qkvh,
                (int)ROWS_X, 1536, 512, 1, 0, 0, 0, 0, 0);
        k_split_heads<<<ew(BHL * HD), 256, 0, stream>>>(qkvh, Qh, Kh, Vt);

        // scores = Q K^T  (batched over B*H), f32 into actF
        gemm_nt(stream, Qh, Kh, nullptr, actF, nullptr,
                LL, LL, HD, BB * HH,
                (long)LL * HD, (long)LL * HD, (long)LL * LL, 0, 0);
        k_softmax<<<dim3(LL, BB * HH), 256, 0, stream>>>(actF, prbH);

        // O = P V  (B of NT-gemm is Vt [HD, L])
        gemm_nt(stream, prbH, Vt, nullptr, nullptr, Oh,
                LL, HD, LL, BB * HH,
                (long)LL * LL, (long)HD * LL, 0, (long)LL * HD, 0);
        k_merge_heads<<<ew(ROWS_X * DD), 256, 0, stream>>>(Oh, Om);

        // out projection -> actF (f32)
        gemm_nt(stream, Om, wOut, F(L_BASE + 12 * l + 3), actF, nullptr,
                (int)ROWS_X, 512, 512, 1, 0, 0, 0, 0, 0);
        k_add_ln<<<dim3((unsigned)ROWS_X), 256, 0, stream>>>(xF, actF,
            F(L_BASE + 12 * l + 4), F(L_BASE + 12 * l + 5), xh);

        // ff1 (exact GELU) -> actH f16 ; ff2 -> actF f32
        gemm_nt(stream, xh, wF1, F(L_BASE + 12 * l + 7), nullptr, actH,
                (int)ROWS_X, 2048, 512, 1, 0, 0, 0, 0, 2);
        gemm_nt(stream, actH, wF2, F(L_BASE + 12 * l + 9), actF, nullptr,
                (int)ROWS_X, 512, 2048, 1, 0, 0, 0, 0, 0);
        k_add_ln<<<dim3((unsigned)ROWS_X), 256, 0, stream>>>(xF, actF,
            F(L_BASE + 12 * l + 10), F(L_BASE + 12 * l + 11), xh);
    }

    // ================= p_sample mean + decoder =================
    k_mean<<<ew(ROWS_X * DD), 256, 0, stream>>>(xtF, xF, NOISE, sched,
        outNP, outNoise, meanH);
    gemm_nt(stream, meanH, whDec, F(P_DEC_B), yPred, nullptr,
            (int)ROWS_X, VV, 512, 1, 0, 0, 0, 0, 0);
}
